// StyleGANv2Generator_77558519431477
// MI455X (gfx1250) — matrix operand order repaired and checked by reading, not compiled, not run on hardware
//
#include <hip/hip_runtime.h>

// ---------------------------------------------------------------------------
// StyleGAN2 generator forward for gfx1250 (MI455X), wave32 + WMMA bf16 + TDM.
// Modulated conv rewritten as: out = d[b,cout] * (w  conv  (s[b,cin]*x)),
// so convs use SHARED bf16 weights -> implicit GEMM with v_wmma_f32_16x16x32_bf16.
// Features are channel-last [b][y][x][c] so each lane's A fragment is two
// contiguous 16B loads; conv weights are tap-major [cout][tap*512+cin] so a
// 32-wide K chunk never crosses a tap. B chunks (32k x 512couts = 32KB) are
// DMA'd into LDS with the Tensor Data Mover, double-buffered, s_wait_tensorcnt.
// ---------------------------------------------------------------------------

typedef __attribute__((ext_vector_type(16))) __bf16 v16bf;
typedef __attribute__((ext_vector_type(8)))  float  v8f;

#define WMMA_BF16(a, b, c) \
  __builtin_amdgcn_wmma_f32_16x16x32_bf16(false, (a), false, (b), (short)0, (c), false, false)

#define NB 32
#define NC 512
#define K9 4608               // 512*9

#if defined(__has_builtin)
#  if __has_builtin(__builtin_amdgcn_tensor_load_to_lds)
#    define HAVE_TDM 1
#  endif
#endif
#ifndef HAVE_TDM
#  define HAVE_TDM 0
#endif

__device__ __forceinline__ float lrelu2(float v) {
  return (v > 0.f ? v : 0.2f * v) * 1.4142135623730951f;
}

#if HAVE_TDM
typedef __attribute__((ext_vector_type(4))) unsigned int v4u_t;
typedef __attribute__((ext_vector_type(8))) int v8i_t;
typedef __attribute__((ext_vector_type(4))) int v4i_t;

// TDM: load a 2D tile (tile_dim0=32 k-elems, tile_dim1=512 couts, row stride
// K9 elems, data_size=2B) from global `gptr` into LDS at `lptr`.
// D# bitfields per CDNA5 ISA 8.3/8.4.
__device__ __forceinline__ void tdm_load_w(const __bf16* gptr, void* lptr) {
  unsigned long long ga = (unsigned long long)(size_t)gptr;
  unsigned int la = (unsigned int)(size_t)lptr;
  v4u_t g0;
  g0[0] = 1u;                                            // count=1, user desc
  g0[1] = la;                                            // lds_addr
  g0[2] = (unsigned int)(ga & 0xFFFFFFFFu);              // global_addr[31:0]
  g0[3] = (unsigned int)((ga >> 32) & 0x1FFFFFFu) | (2u << 30);  // [56:32] | type=2
  v8i_t g1;
  g1[0] = 0x00010000;                      // wg_mask=0, data_size=1 (2 bytes)
  g1[1] = (int)((K9 & 0xFFFF) << 16);      // tensor_dim0[15:0] @ [31:16]
  g1[2] = (int)((K9 >> 16) | ((NC & 0xFFFF) << 16));  // dim0 hi | tensor_dim1 lo
  g1[3] = (int)((NC >> 16) | (32 << 16));  // tensor_dim1 hi | tile_dim0=32
  g1[4] = NC;                              // tile_dim1=512, tile_dim2=0
  g1[5] = K9;                              // tensor_dim0_stride[31:0]
  g1[6] = 0;                               // stride hi | dim1_stride lo
  g1[7] = 0;
  v4i_t z4 = {0, 0, 0, 0};
#if __clang_major__ >= 23
  v8i_t z8 = {0, 0, 0, 0, 0, 0, 0, 0};
  __builtin_amdgcn_tensor_load_to_lds(g0, g1, z4, z4, z8, 0);
#else
  __builtin_amdgcn_tensor_load_to_lds(g0, g1, z4, z4, 0);
#endif
}
#endif  // HAVE_TDM

// ---------------------------------------------------------------------------
// Mapping network: pixel_norm + 8 x [x = lrelu(x @ (W^T/sqrt512) + b*0.01)*sqrt2]
// One workgroup (8 waves). x lives in LDS as bf16. 64 WMMA tiles per layer.
// ---------------------------------------------------------------------------
__global__ __launch_bounds__(256) void map_mlp(const float* __restrict__ z,
                                               const float* __restrict__ mlpw,
                                               const float* __restrict__ mlpb,
                                               float* __restrict__ wout) {
  __shared__ __bf16 xh[NB][NC];   // 32 KB
  __shared__ float  rn[NB];
  int tid = threadIdx.x;
  if (tid < NB) {
    float s = 0.f;
    for (int k = 0; k < NC; ++k) { float v = z[tid * NC + k]; s += v * v; }
    rn[tid] = 1.f / (sqrtf(s) * 0.044194173824159216f + 1e-6f);  // /sqrt(512)
  }
  __syncthreads();
  for (int e = tid; e < NB * NC; e += 256) {
    int b = e >> 9, k = e & 511;
    xh[b][k] = (__bf16)(z[e] * rn[b]);
  }
  __syncthreads();

  int lane = tid & 31, wave = tid >> 5;
  int row = lane & 15, half = lane >> 4;
  const float sc = 0.044194173824159216f;  // 1/sqrt(512)

  for (int l = 0; l < 8; ++l) {
    const float* Wl = mlpw + (size_t)l * NC * NC;
    const float* Bl = mlpb + l * NC;
    float outv[8][8];
    for (int t = 0; t < 8; ++t) {
      int id = t * 8 + wave;          // 64 tiles: 2 M-tiles x 32 N-tiles
      int nt = id >> 1, mt = id & 1;
      v8f acc = {};
      for (int kc = 0; kc < 16; ++kc) {
        union { v16bf v; __bf16 e[16]; } A, Bm;
#pragma unroll
        for (int i = 0; i < 16; ++i) {           // A 16x32 bf16 layout
          int k = kc * 32 + half * 16 + i;
          A.e[i] = xh[mt * 16 + row][k];
        }
#pragma unroll
        for (int i = 0; i < 16; ++i) {           // B 32x16 bf16 layout
          int k = kc * 32 + half * 16 + i;
          Bm.e[i] = (__bf16)(Wl[(size_t)(nt * 16 + row) * NC + k] * sc);
        }
        acc = WMMA_BF16(A.v, Bm.v, acc);
      }
#pragma unroll
      for (int r = 0; r < 8; ++r)
        outv[t][r] = lrelu2(acc[r] + Bl[nt * 16 + row] * 0.01f);
    }
    __syncthreads();
    for (int t = 0; t < 8; ++t) {
      int id = t * 8 + wave;
      int nt = id >> 1, mt = id & 1;
#pragma unroll
      for (int r = 0; r < 8; ++r) {              // C f32 layout: row=half*8+r
        int gm = mt * 16 + half * 8 + r;
        int gn = nt * 16 + row;
        xh[gm][gn] = (__bf16)outv[t][r];
        if (l == 7) wout[gm * NC + gn] = outv[t][r];
      }
    }
    __syncthreads();
  }
}

// ---------------------------------------------------------------------------
// Generic (32x512) = (32x512) @ (512x512)^T GEMM, WMMA bf16.
// mode 0: out = A@Wg^T*scale + bias          (style modulation)
// mode 1: out = rsqrt((A*A)@Wg^T*scale+1e-8) (demodulation factors)
// ---------------------------------------------------------------------------
__global__ __launch_bounds__(256) void gemm_bc(const float* __restrict__ Amat,
                                               const float* __restrict__ Wg,
                                               const float* __restrict__ bias,
                                               float* __restrict__ outp,
                                               int mode, float scale) {
  int lane = threadIdx.x & 31, wave = threadIdx.x >> 5;
  int id = blockIdx.x * 8 + wave;
  int nt = id >> 1, mt = id & 1;
  int row = lane & 15, half = lane >> 4;
  v8f acc = {};
  for (int kc = 0; kc < 16; ++kc) {
    union { v16bf v; __bf16 e[16]; } A, Bm;
#pragma unroll
    for (int i = 0; i < 16; ++i) {
      int k = kc * 32 + half * 16 + i;
      float v = Amat[(mt * 16 + row) * NC + k];
      if (mode == 1) v *= v;
      A.e[i] = (__bf16)v;
    }
#pragma unroll
    for (int i = 0; i < 16; ++i) {
      int k = kc * 32 + half * 16 + i;
      Bm.e[i] = (__bf16)(Wg[(size_t)(nt * 16 + row) * NC + k] * scale);
    }
    acc = WMMA_BF16(A.v, Bm.v, acc);
  }
#pragma unroll
  for (int r = 0; r < 8; ++r) {
    int gm = mt * 16 + half * 8 + r;
    int gn = nt * 16 + row;
    float v = acc[r];
    v = (mode == 0) ? (v + bias[gn]) : rsqrtf(v + 1e-8f);
    outp[gm * NC + gn] = v;
  }
}

// W2[cout*512+cin] = sum over 3x3 of w^2
__global__ void w2_k(const float* __restrict__ w, float* __restrict__ W2) {
  int idx = blockIdx.x * 256 + threadIdx.x;
  if (idx >= NC * NC) return;
  const float* p = w + (size_t)idx * 9;
  float s = 0.f;
#pragma unroll
  for (int j = 0; j < 9; ++j) s += p[j] * p[j];
  W2[idx] = s;
}

// Reorder conv weights (Cout,Cin,3,3) f32 -> tap-major bf16 [cout][tap*512+cin]
__global__ void wrord_k(const float* __restrict__ w, __bf16* __restrict__ d) {
  size_t i = (size_t)blockIdx.x * 256 + threadIdx.x;
  if (i >= (size_t)NC * K9) return;
  int cout = (int)(i / K9);
  int rem = (int)(i % K9);
  int r = rem / NC, cin = rem - r * NC;
  d[i] = (__bf16)w[((size_t)cout * NC + cin) * 9 + r];
}

// Tile const_input (1,512,4,4) planar f32 -> channel-last bf16 [b][p][c]
__global__ void tile_const_k(const float* __restrict__ cst, __bf16* __restrict__ d) {
  int i = blockIdx.x * 256 + threadIdx.x;
  if (i >= NB * 16 * NC) return;
  int c = i & (NC - 1);
  int p = (i / NC) & 15;
  d[i] = (__bf16)cst[c * 16 + p];
}

// xmod[b,p,c] = bf16( x[b,p,c] * s[b,c] )   (channel-last, lane=c coalesced)
__global__ void modx_k(const __bf16* __restrict__ xin, const float* __restrict__ s,
                       __bf16* __restrict__ xm, int RR) {
  size_t i = (size_t)blockIdx.x * 256 + threadIdx.x;
  size_t tot = (size_t)NB * NC * RR;
  if (i >= tot) return;
  int c = (int)(i & (NC - 1));
  int b = (int)(i / ((size_t)NC * RR));
  xm[i] = (__bf16)((float)xin[i] * s[b * NC + c]);
}

// ---------------------------------------------------------------------------
// 3x3 conv, implicit GEMM: M = 16 pixels (4x4 tile), N = cout, K = tap*512+cin.
// Grid (R*R/16, NB), 8 waves; each wave owns 4 N-tiles, A chunk reused 4x.
// B chunk (32KB) staged in LDS via TDM, double-buffered.
// Epilogue: *1/sqrt(4608) * d[b,cout] + noise, +bias, leaky*sqrt2 -> bf16.
// ---------------------------------------------------------------------------
__global__ __launch_bounds__(256) void conv3x3_wmma(
    const __bf16* __restrict__ xmod, const __bf16* __restrict__ wbf,
    const float* __restrict__ dmod, const float* __restrict__ noise,
    const float* __restrict__ nw, const float* __restrict__ abias,
    __bf16* __restrict__ xout, int R) {
#if HAVE_TDM
  __shared__ __bf16 bsh[2][NC * 32];  // 2 x 32 KB double buffer
#endif
  int b = blockIdx.y;
  int tile = blockIdx.x;
  int tpr = R >> 2;
  int ty = (tile / tpr) << 2, tx = (tile % tpr) << 2;
  int lane = threadIdx.x & 31, wave = threadIdx.x >> 5;
  int row = lane & 15, half = lane >> 4;
  int py = ty + (row >> 2), px = tx + (row & 3);
  const __bf16* xb = xmod + (size_t)b * R * R * NC;
  v8f acc[4] = {};
#if HAVE_TDM
  if (wave == 0) tdm_load_w(wbf, (void*)&bsh[0][0]);
#endif
  for (int r9 = 0; r9 < 9; ++r9) {
    int dy = r9 / 3, dx = r9 - dy * 3;
    int yy = py + dy - 1, xx = px + dx - 1;
    bool val = (yy >= 0 && yy < R && xx >= 0 && xx < R);
    const __bf16* xpix = xb + ((size_t)yy * R + xx) * NC;
    for (int c16 = 0; c16 < 16; ++c16) {
      int kc = r9 * 16 + c16;
      union { v16bf v; __bf16 e[16]; } A;
      const __bf16* p1 = xpix + c16 * 32 + half * 16;
#pragma unroll
      for (int i = 0; i < 8; ++i) {          // two contiguous 16B runs per lane
        A.e[i]     = val ? p1[i]      : (__bf16)0.f;
        A.e[8 + i] = val ? p1[8 + i]  : (__bf16)0.f;
      }
#if HAVE_TDM
      __syncthreads();                       // prev compute done -> prefetch safe
      if (wave == 0) {
        if (kc + 1 < 144) {
          tdm_load_w(wbf + (size_t)(kc + 1) * 32, (void*)&bsh[(kc + 1) & 1][0]);
          __builtin_amdgcn_s_wait_tensorcnt(1);   // chunk kc done (in-order)
        } else {
          __builtin_amdgcn_s_wait_tensorcnt(0);
        }
      }
      __syncthreads();                       // announce buf[kc&1] ready
      const __bf16* bsrc = &bsh[kc & 1][0];
#endif
#pragma unroll
      for (int nt = 0; nt < 4; ++nt) {
        int cout = (wave * 4 + nt) * 16 + row;
        union { v16bf v; __bf16 e[16]; } Bm;
#if HAVE_TDM
        const __bf16* wp = bsrc + cout * 32 + half * 16;
#else
        const __bf16* wp = wbf + (size_t)cout * K9 + kc * 32 + half * 16;
#endif
#pragma unroll
        for (int i = 0; i < 16; ++i) Bm.e[i] = wp[i];
        acc[nt] = WMMA_BF16(A.v, Bm.v, acc[nt]);
      }
    }
  }
  float nwv = nw[0];
#pragma unroll
  for (int nt = 0; nt < 4; ++nt) {
    int cout = (wave * 4 + nt) * 16 + row;
    float dm = dmod[b * NC + cout] * 0.014731391274719739f;  // 1/sqrt(4608)
#pragma unroll
    for (int r = 0; r < 8; ++r) {
      int pm = half * 8 + r;
      int oy = ty + (pm >> 2), ox = tx + (pm & 3);
      float v = acc[nt][r] * dm + nwv * noise[((size_t)b * R + oy) * R + ox];
      v = lrelu2(v + abias[cout]);
      xout[(((size_t)b * R + oy) * R + ox) * NC + cout] = (__bf16)v;
    }
  }
}

// ---------------------------------------------------------------------------
// Transposed conv (stride 2, pad 0): out[2yi+dy, 2xi+dx] += x[yi,xi]*w[dy,dx].
// Same implicit GEMM; tap validity parity-masked per chunk. Ro = 2R+1.
// Writes demodulated pre-blur result (channel-last bf16).
// ---------------------------------------------------------------------------
__global__ __launch_bounds__(256) void convT_wmma(
    const __bf16* __restrict__ xmod, const __bf16* __restrict__ wbf,
    const float* __restrict__ dmod, __bf16* __restrict__ tout, int R) {
#if HAVE_TDM
  __shared__ __bf16 bsh[2][NC * 32];
#endif
  int Ro = 2 * R + 1;
  int npix = Ro * Ro;
  int b = blockIdx.y;
  int tile = blockIdx.x;
  int lane = threadIdx.x & 31, wave = threadIdx.x >> 5;
  int row = lane & 15, half = lane >> 4;
  int pm_in = tile * 16 + row;
  bool mvalid = pm_in < npix;
  int py = mvalid ? pm_in / Ro : 0;
  int px = mvalid ? pm_in % Ro : 0;
  const __bf16* xb = xmod + (size_t)b * R * R * NC;
  v8f acc[4] = {};
#if HAVE_TDM
  if (wave == 0) tdm_load_w(wbf, (void*)&bsh[0][0]);
#endif
  for (int r9 = 0; r9 < 9; ++r9) {
    int dy = r9 / 3, dx = r9 - dy * 3;
    int yy = py - dy, xx = px - dx;
    bool val = mvalid && yy >= 0 && xx >= 0 && !(yy & 1) && !(xx & 1) &&
               (yy >> 1) < R && (xx >> 1) < R;
    const __bf16* xpix = xb + ((size_t)(yy >> 1) * R + (xx >> 1)) * NC;
    for (int c16 = 0; c16 < 16; ++c16) {
      int kc = r9 * 16 + c16;
      union { v16bf v; __bf16 e[16]; } A;
      const __bf16* p1 = xpix + c16 * 32 + half * 16;
#pragma unroll
      for (int i = 0; i < 8; ++i) {
        A.e[i]     = val ? p1[i]      : (__bf16)0.f;
        A.e[8 + i] = val ? p1[8 + i]  : (__bf16)0.f;
      }
#if HAVE_TDM
      __syncthreads();
      if (wave == 0) {
        if (kc + 1 < 144) {
          tdm_load_w(wbf + (size_t)(kc + 1) * 32, (void*)&bsh[(kc + 1) & 1][0]);
          __builtin_amdgcn_s_wait_tensorcnt(1);
        } else {
          __builtin_amdgcn_s_wait_tensorcnt(0);
        }
      }
      __syncthreads();
      const __bf16* bsrc = &bsh[kc & 1][0];
#endif
#pragma unroll
      for (int nt = 0; nt < 4; ++nt) {
        int cout = (wave * 4 + nt) * 16 + row;
        union { v16bf v; __bf16 e[16]; } Bm;
#if HAVE_TDM
        const __bf16* wp = bsrc + cout * 32 + half * 16;
#else
        const __bf16* wp = wbf + (size_t)cout * K9 + kc * 32 + half * 16;
#endif
#pragma unroll
        for (int i = 0; i < 16; ++i) Bm.e[i] = wp[i];
        acc[nt] = WMMA_BF16(A.v, Bm.v, acc[nt]);
      }
    }
  }
#pragma unroll
  for (int nt = 0; nt < 4; ++nt) {
    int cout = (wave * 4 + nt) * 16 + row;
    float dm = dmod[b * NC + cout] * 0.014731391274719739f;
#pragma unroll
    for (int r = 0; r < 8; ++r) {
      int pm = tile * 16 + half * 8 + r;
      if (pm < npix) {
        int oy = pm / Ro, ox = pm % Ro;
        tout[(((size_t)b * Ro + oy) * Ro + ox) * NC + cout] = (__bf16)(acc[nt][r] * dm);
      }
    }
  }
}

// Blur (4x4, *4/64) of Ro=(S+1) map -> S, fused noise + bias + act -> bf16.
// Channel-last: lane = channel -> every tap is a coalesced 64B wave read.
__global__ void blur_act(const __bf16* __restrict__ tin, const float* __restrict__ noise,
                         const float* __restrict__ nw, const float* __restrict__ abias,
                         __bf16* __restrict__ xout, int S) {
  int Ro = S + 1;
  size_t idx = (size_t)blockIdx.x * 256 + threadIdx.x;
  size_t total = (size_t)NB * NC * S * S;
  if (idx >= total) return;
  int c = (int)(idx & (NC - 1));
  int x = (int)((idx / NC) % S);
  int y = (int)((idx / ((size_t)NC * S)) % S);
  int b = (int)(idx / ((size_t)NC * S * S));
  const float kv[4] = {1.f, 3.f, 3.f, 1.f};
  float s = 0.f;
#pragma unroll
  for (int a = 0; a < 4; ++a) {
    int yy = y + a - 1;
    if (yy < 0 || yy >= Ro) continue;
#pragma unroll
    for (int e = 0; e < 4; ++e) {
      int xx = x + e - 1;
      if (xx < 0 || xx >= Ro) continue;
      s += kv[a] * kv[e] * (float)tin[(((size_t)b * Ro + yy) * Ro + xx) * NC + c];
    }
  }
  float v = s * (1.f / 16.f) + nw[0] * noise[((size_t)b * S + y) * S + x];
  v = lrelu2(v + abias[c]);
  xout[idx] = (__bf16)v;
}

// toRGB 1x1 conv (no demod): dst[b,c3,p] (+)= sum_cin w*s*x / sqrt(512) + bias
__global__ void torgb_k(const __bf16* __restrict__ xin, const float* __restrict__ s,
                        const float* __restrict__ wrgb, const float* __restrict__ tb,
                        float* __restrict__ dst, int R, int addf) {
  int RR = R * R;
  size_t idx = (size_t)blockIdx.x * 256 + threadIdx.x;
  size_t total = (size_t)NB * 3 * RR;
  if (idx >= total) return;
  int p = (int)(idx % RR);
  int c3 = (int)((idx / RR) % 3);
  int b = (int)(idx / ((size_t)3 * RR));
  const __bf16* xp = xin + ((size_t)b * RR + p) * NC;  // channel-last: contiguous
  float acc = 0.f;
  for (int cin = 0; cin < NC; ++cin)
    acc += wrgb[c3 * NC + cin] * s[b * NC + cin] * (float)xp[cin];
  float v = acc * 0.044194173824159216f + tb[c3];
  if (addf) dst[idx] += v; else dst[idx] = v;
}

// Skip upsample: upfirdn2d(skip, 4*blur, up=2, pad0=2, pad1=1):
// out[q] = sum_a k[a] * u[q+a-2], u = zero-stuffed skip.
__global__ void skip_up_k(const float* __restrict__ sin, float* __restrict__ dst, int Rin) {
  int S = 2 * Rin;
  size_t idx = (size_t)blockIdx.x * 256 + threadIdx.x;
  size_t total = (size_t)NB * 3 * S * S;
  if (idx >= total) return;
  int x = (int)(idx % S);
  int y = (int)((idx / S) % S);
  int c = (int)((idx / ((size_t)S * S)) % 3);
  int b = (int)(idx / ((size_t)S * S * 3));
  const float* sp = sin + ((size_t)b * 3 + c) * Rin * Rin;
  const float kv[4] = {1.f, 3.f, 3.f, 1.f};
  float acc = 0.f;
#pragma unroll
  for (int a = 0; a < 4; ++a) {
    int qy = y + a - 2;
    if (qy < 0 || (qy & 1)) continue;
    int iy = qy >> 1;
    if (iy >= Rin) continue;
#pragma unroll
    for (int e = 0; e < 4; ++e) {
      int qx = x + e - 2;
      if (qx < 0 || (qx & 1)) continue;
      int ix = qx >> 1;
      if (ix >= Rin) continue;
      acc += kv[a] * kv[e] * sp[iy * Rin + ix];
    }
  }
  dst[idx] = acc * (1.f / 16.f);
}

// ---------------------------------------------------------------------------
extern "C" void kernel_launch(void* const* d_in, const int* in_sizes, int n_in,
                              void* d_out, int out_size, void* d_ws, size_t ws_size,
                              hipStream_t stream) {
  (void)in_sizes; (void)n_in; (void)out_size; (void)ws_size;
  const float* z     = (const float*)d_in[0];
  const float* mlpw  = (const float*)d_in[1];
  const float* mlpb  = (const float*)d_in[2];
  const float* cst   = (const float*)d_in[3];
  const float* convw = (const float*)d_in[4];
  const float* csw   = (const float*)d_in[5];
  const float* csb   = (const float*)d_in[6];
  const float* nw    = (const float*)d_in[7];
  const float* ab    = (const float*)d_in[8];
  const float* tw    = (const float*)d_in[9];
  const float* tsw   = (const float*)d_in[10];
  const float* tsb   = (const float*)d_in[11];
  const float* tbias = (const float*)d_in[12];
  const float* noise[9];
  for (int i = 0; i < 9; ++i) noise[i] = (const float*)d_in[13 + i];

  char* ws = (char*)d_ws;
  float*  Wlat = (float*)(ws + ((size_t)0 << 20));
  float*  Sbuf = (float*)(ws + ((size_t)1 << 20));
  float*  Dbuf = (float*)(ws + ((size_t)2 << 20));
  float*  W2   = (float*)(ws + ((size_t)3 << 20));
  __bf16* wbf  = (__bf16*)(ws + ((size_t)4 << 20));     // 4.5 MB
  __bf16* xA   = (__bf16*)(ws + ((size_t)16 << 20));    // 128 MB
  __bf16* xB   = (__bf16*)(ws + ((size_t)160 << 20));   // 128 MB
  __bf16* xM   = (__bf16*)(ws + ((size_t)304 << 20));   // 128 MB
  __bf16* Tb   = (__bf16*)(ws + ((size_t)448 << 20));   // ~133 MB
  float*  skA  = (float*)(ws + ((size_t)592 << 20));
  float*  skB  = (float*)(ws + ((size_t)596 << 20));

  const float SC512  = 0.044194173824159216f;  // 1/sqrt(512)
  const float SC4608 = 1.f / 4608.f;           // scale^2 for demod

  map_mlp<<<1, 256, 0, stream>>>(z, mlpw, mlpb, Wlat);
  tile_const_k<<<(NB * NC * 16 + 255) / 256, 256, 0, stream>>>(cst, xA);

  __bf16* cur = xA;
  __bf16* nxt = xB;
  int R = 4;

  auto conv_common = [&](int li) {
    gemm_bc<<<8, 256, 0, stream>>>(Wlat, csw + (size_t)li * NC * NC, csb + li * NC,
                                   Sbuf, 0, SC512);
    w2_k<<<(NC * NC + 255) / 256, 256, 0, stream>>>(convw + (size_t)li * NC * NC * 9, W2);
    gemm_bc<<<8, 256, 0, stream>>>(Sbuf, W2, csb, Dbuf, 1, SC4608);
    wrord_k<<<((NC * K9) + 255) / 256, 256, 0, stream>>>(
        convw + (size_t)li * NC * NC * 9, wbf);
    size_t tot = (size_t)NB * NC * R * R;
    modx_k<<<(unsigned)((tot + 255) / 256), 256, 0, stream>>>(cur, Sbuf, xM, R * R);
  };

  // layer 0 (res 4, no up)
  conv_common(0);
  conv3x3_wmma<<<dim3(R * R / 16, NB), 256, 0, stream>>>(xM, wbf, Dbuf, noise[0],
                                                         nw + 0, ab + 0, nxt, R);
  { __bf16* t = cur; cur = nxt; nxt = t; }

  // toRGB 0 (res 4)
  gemm_bc<<<8, 256, 0, stream>>>(Wlat, tsw, tsb, Sbuf, 0, SC512);
  torgb_k<<<(unsigned)(((size_t)NB * 3 * R * R + 255) / 256), 256, 0, stream>>>(
      cur, Sbuf, tw, tbias, skA, R, 0);

  for (int r = 0; r < 4; ++r) {
    // --- up layer ---
    int li = 1 + 2 * r;
    conv_common(li);                       // modx at input res R
    int Ro = 2 * R + 1;
    convT_wmma<<<dim3((Ro * Ro + 15) / 16, NB), 256, 0, stream>>>(xM, wbf, Dbuf, Tb, R);
    int S = 2 * R;
    blur_act<<<(unsigned)(((size_t)NB * NC * S * S + 255) / 256), 256, 0, stream>>>(
        Tb, noise[li], nw + li, ab + (size_t)li * NC, nxt, S);
    { __bf16* t = cur; cur = nxt; nxt = t; }
    R = S;

    // --- plain layer ---
    li = 2 + 2 * r;
    conv_common(li);
    conv3x3_wmma<<<dim3(R * R / 16, NB), 256, 0, stream>>>(
        xM, wbf, Dbuf, noise[li], nw + li, ab + (size_t)li * NC, nxt, R);
    { __bf16* t = cur; cur = nxt; nxt = t; }

    // --- skip path at res R ---
    float* dst = (r == 3) ? (float*)d_out : skB;
    skip_up_k<<<(unsigned)(((size_t)NB * 3 * R * R + 255) / 256), 256, 0, stream>>>(
        skA, dst, R / 2);
    int j = r + 1;
    gemm_bc<<<8, 256, 0, stream>>>(Wlat, tsw + (size_t)j * NC * NC, tsb + j * NC,
                                   Sbuf, 0, SC512);
    torgb_k<<<(unsigned)(((size_t)NB * 3 * R * R + 255) / 256), 256, 0, stream>>>(
        cur, Sbuf, tw + (size_t)j * 3 * NC, tbias + j * 3, dst, R, 1);
    if (r < 3) { float* t = skA; skA = skB; skB = t; }
  }
}
